// FourierStepHFCFilter_9732395893275
// MI455X (gfx1250) — matrix operand order repaired and checked by reading, not compiled, not run on hardware
//
#include <hip/hip_runtime.h>
#include <math.h>

typedef unsigned int u32;
typedef unsigned short u16;

typedef __attribute__((ext_vector_type(8)))  float          v8f;
typedef __attribute__((ext_vector_type(8)))  unsigned short v8us;
typedef __attribute__((ext_vector_type(16))) __bf16         v16bf;

union Frag { v8us h[2]; v16bf bf; };

#define NSZ      512
#define NPIX     262144      // 512*512
#define NCH      96          // B*C
#define NTOT     25165824    // NCH*NPIX
#define MED_RANK 131071      // (NPIX-1)//2
#define PBINS    16384

// ---- workspace layout (bytes) ----
static constexpr size_t OFF_KHI  = 0;                                   // bf16 [512*512]
static constexpr size_t OFF_KLO  = 524288;
static constexpr size_t OFF_K2HI = 1048576;
static constexpr size_t OFF_K2LO = 1572864;
static constexpr size_t OFF_XHI  = 2097152;                             // bf16 [96*512*512]
static constexpr size_t OFF_XLO  = OFF_XHI + (size_t)NCH * NPIX * 2;
static constexpr size_t OFF_YHI  = OFF_XLO + (size_t)NCH * NPIX * 2;
static constexpr size_t OFF_YLO  = OFF_YHI + (size_t)NCH * NPIX * 2;
static constexpr size_t OFF_KU   = OFF_YLO + (size_t)NCH * NPIX * 2;    // double[512]
static constexpr size_t OFF_KV   = OFF_KU + 4096;
static constexpr size_t OFF_MED  = OFF_KV + 4096;                       // float[96]
static constexpr size_t OFF_BKT  = OFF_MED + 512;                       // u32[96]
static constexpr size_t OFF_REM  = OFF_BKT + 512;
static constexpr size_t OFF_PLO  = OFF_REM + 512;
static constexpr size_t OFF_PHI  = OFF_PLO + 512;
static constexpr size_t OFF_HIST = OFF_PHI + 512;                       // u32[96*65536]

// ---- helpers ----
__device__ inline u16 f2bf(float f) {
    u32 u = __float_as_uint(f);
    u += 0x7FFFu + ((u >> 16) & 1u);       // round to nearest even
    return (u16)(u >> 16);
}
__device__ inline void split2(float f, u16& hh, u16& ll) {
    hh = f2bf(f);
    float fh = __uint_as_float(((u32)hh) << 16);
    ll = f2bf(f - fh);
}
__device__ inline u32 f2key(float f) {
    u32 u = __float_as_uint(f);
    return (u & 0x80000000u) ? ~u : (u | 0x80000000u);
}
__device__ inline float key2f(u32 key) {
    u32 u = (key & 0x80000000u) ? (key ^ 0x80000000u) : ~key;
    return __uint_as_float(u);
}
__device__ inline v8f wmma_bf16(v16bf a, v16bf b, v8f c) {
    return __builtin_amdgcn_wmma_f32_16x16x32_bf16(false, a, false, b, (short)0, c, false, false);
}
// 32-bit LDS byte offset of a __shared__ object (for async-to-LDS vdst operand)
__device__ inline unsigned lds_off(void* p) {
    return (unsigned)(unsigned long long)(__attribute__((address_space(3))) char*)p;
}
// CDNA5 async copy: global -> LDS, 16 bytes, tracked by ASYNCcnt (no VGPR round-trip)
__device__ inline void async_load_b128(unsigned lds_byte_off, const void* gptr) {
    asm volatile("global_load_async_to_lds_b128 %0, %1, off"
                 :: "v"(lds_byte_off), "v"(gptr) : "memory");
}
__device__ inline void wait_asynccnt0() {
    asm volatile("s_wait_asynccnt 0x0" ::: "memory");
}

// ---- kernels ----
__global__ void k_zero(u32* __restrict__ p, int n) {
    int i = blockIdx.x * 256 + threadIdx.x;
    if (i < n) p[i] = 0;
}

// Dirichlet kernels from the actual filter_map (separable: fu = fmap[:,0], fv = fmap[0,:])
__global__ void k_build_kd(const float* __restrict__ fmap, double* __restrict__ ku, double* __restrict__ kv) {
    int t = blockIdx.x * 512 + threadIdx.x;   // <<<2,512>>>
    if (t >= 1024) return;
    int d = t & 511;
    bool isV = t >= 512;
    double s = 0.0;
    for (int u = 0; u < 512; ++u) {
        float f = isV ? fmap[u] : fmap[(size_t)u * NSZ];
        if (f != 0.0f) {
            int ph = (u * d) & 511;
            s += (double)f * cos(6.283185307179586476925287 * (double)ph / 512.0);
        }
    }
    (isV ? kv : ku)[d] = s / 512.0;
}

__global__ void k_build_K(const double* __restrict__ ku, const double* __restrict__ kv,
                          u16* __restrict__ Khi, u16* __restrict__ Klo,
                          u16* __restrict__ K2hi, u16* __restrict__ K2lo) {
    int i = blockIdx.x * 256 + threadIdx.x;
    if (i >= NPIX) return;
    int r = i >> 9, c = i & 511;
    float kl = (float)ku[(r - c) & 511];   // K_left[m][k]  = ku[(m-k) mod 512]
    float kr = (float)kv[(c - r) & 511];   // K_right[k][n] = kv[(n-k) mod 512]
    split2(kl, Khi[i], Klo[i]);
    split2(kr, K2hi[i], K2lo[i]);
}

// -- median radix-select, pass 1: histogram of top 16 key bits
__global__ void k_hist_hi(const float* __restrict__ x, u32* __restrict__ hist) {
    int i = blockIdx.x * 256 + threadIdx.x;
    if (i >= NTOT) return;
    int c = i >> 18;
    u32 key = f2key(x[i]);
    atomicAdd(&hist[((size_t)c << 16) + (key >> 16)], 1u);
}

__global__ void k_select1(u32* __restrict__ hist, u32* __restrict__ bucket, u32* __restrict__ rem) {
    int c = blockIdx.x, tid = threadIdx.x;
    __shared__ u32 part[256];
    u32* h = hist + ((size_t)c << 16);
    u32 s = 0;
    for (int i = 0; i < 256; ++i) s += h[tid * 256 + i];
    part[tid] = s;
    __syncthreads();
    if (tid == 0) {
        u32 k = MED_RANK, cum = 0;
        int cc = 0;
        for (; cc < 256; ++cc) { u32 p = part[cc]; if (cum + p > k) break; cum += p; }
        for (int i = 0; i < 256; ++i) {
            u32 v = h[cc * 256 + i];
            if (cum + v > k) { bucket[c] = (u32)(cc * 256 + i); rem[c] = k - cum; break; }
            cum += v;
        }
    }
    __syncthreads();
    for (int i = 0; i < 256; ++i) h[tid * 256 + i] = 0;
}

// -- median pass 2: histogram of low 16 bits within the selected bucket
__global__ void k_hist_lo(const float* __restrict__ x, const u32* __restrict__ bucket, u32* __restrict__ hist) {
    int i = blockIdx.x * 256 + threadIdx.x;
    if (i >= NTOT) return;
    int c = i >> 18;
    u32 key = f2key(x[i]);
    if ((key >> 16) == bucket[c])
        atomicAdd(&hist[((size_t)c << 16) + (key & 0xFFFFu)], 1u);
}

__global__ void k_select2(u32* __restrict__ hist, const u32* __restrict__ bucket,
                          const u32* __restrict__ rem, float* __restrict__ med) {
    int c = blockIdx.x, tid = threadIdx.x;
    __shared__ u32 part[256];
    u32* h = hist + ((size_t)c << 16);
    u32 s = 0;
    for (int i = 0; i < 256; ++i) s += h[tid * 256 + i];
    part[tid] = s;
    __syncthreads();
    if (tid == 0) {
        u32 k = rem[c], cum = 0;
        int cc = 0;
        for (; cc < 256; ++cc) { u32 p = part[cc]; if (cum + p > k) break; cum += p; }
        for (int i = 0; i < 256; ++i) {
            u32 v = h[cc * 256 + i];
            if (cum + v > k) {
                u32 key = (bucket[c] << 16) | (u32)(cc * 256 + i);
                med[c] = key2f(key) + 0.2f;
                break;
            }
            cum += v;
        }
    }
    __syncthreads();
    for (int i = 0; i < 256; ++i) h[tid * 256 + i] = 0;   // leaves hist zeroed for pct pass
}

// -- median padding + bf16 hi/lo split of X
__global__ void k_split_x(const float* __restrict__ x, const float* __restrict__ mask,
                          const float* __restrict__ med, u16* __restrict__ Xhi, u16* __restrict__ Xlo) {
    int i = blockIdx.x * 256 + threadIdx.x;
    if (i >= NTOT) return;
    int c = i >> 18;
    int b = i / 786432;           // 3*NPIX
    int pix = i & (NPIX - 1);
    float m = mask[(size_t)b * NPIX + pix];
    float xp = m * x[i] + (1.0f - m) * med[c];
    split2(xp, Xhi[i], Xlo[i]);
}

// -- bf16x3 split GEMM: out[m][n] = sum_k A[m][k]*B[k][n], 512x512x512 per channel.
//    Block tile 128(M) x 64(N), k-step 32; 8 waves, each owns a 32x32 subtile (4 WMMA tiles).
//    A panel staged with CDNA5 async global->LDS copies (ASYNCcnt); B panel staged with a
//    k-transpose scatter so each lane's 16 B elements are two 8-wide k runs (2x ds_load_b128).
//    mode 0: store split-bf16 (feeds pass 2);  mode 1: store fabs(f32) to out.
__global__ __launch_bounds__(256) void k_gemm(
        const u16* __restrict__ Ahi, const u16* __restrict__ Alo, size_t aStride,
        const u16* __restrict__ Bhi, const u16* __restrict__ Blo, size_t bStride,
        u16* __restrict__ oHi, u16* __restrict__ oLo, float* __restrict__ oF, int mode) {
    __shared__ __align__(16) u16 sAh[128 * 32], sAl[128 * 32];
    __shared__ __align__(16) u16 sBh[64 * 32],  sBl[64 * 32];   // k-transposed: [n_local][k_local]

    int bid = blockIdx.x;
    int ch = bid >> 5;
    int t = bid & 31;
    int m0 = (t >> 3) * 128;
    int n0 = (t & 7) * 64;

    const u16* Ah = Ahi + (size_t)ch * aStride;
    const u16* Al = Alo + (size_t)ch * aStride;
    const u16* Bh = Bhi + (size_t)ch * bStride;
    const u16* Bl = Blo + (size_t)ch * bStride;

    int tid = threadIdx.x;
    int wave = tid >> 5, lane = tid & 31;
    int wm = (wave & 3) << 5;     // 0,32,64,96
    int wn = (wave >> 2) << 5;    // 0,32
    int half = lane >> 4, mr = lane & 15;

    v8f acc[2][2] = {};

    for (int kb = 0; kb < NSZ; kb += 32) {
        __syncthreads();
        // stage A panel 128x32 (hi & lo) via async copies: 512 x 16B chunks per buffer
        for (int idx = tid; idx < 512; idx += 256) {
            int r = idx >> 2, sseg = idx & 3;
            size_t g = (size_t)(m0 + r) * NSZ + kb + (sseg << 3);
            unsigned l = (unsigned)(((r << 5) + (sseg << 3)) * 2);   // byte offset in panel
            async_load_b128(lds_off(sAh) + l, &Ah[g]);
            async_load_b128(lds_off(sAl) + l, &Al[g]);
            if (kb + 32 < NSZ) {
                __builtin_prefetch(&Ah[g + 32], 0, 0);
                __builtin_prefetch(&Al[g + 32], 0, 0);
            }
        }
        // stage B panel 32x64, transposed into [n][k]
        for (int idx = tid; idx < 512; idx += 256) {
            int k = idx >> 4, g4 = idx & 15;
            size_t g = (size_t)(kb + k) * NSZ + n0 + (g4 << 2);
            uint2 hh = *(const uint2*)&Bh[g];
            uint2 ll = *(const uint2*)&Bl[g];
            const u16* hp = (const u16*)&hh;
            const u16* lp = (const u16*)&ll;
#pragma unroll
            for (int j = 0; j < 4; ++j) {
                sBh[((g4 << 2) + j) * 32 + k] = hp[j];
                sBl[((g4 << 2) + j) * 32 + k] = lp[j];
            }
            if (kb + 32 < NSZ) {
                __builtin_prefetch(&Bh[g + (size_t)32 * NSZ], 0, 0);
                __builtin_prefetch(&Bl[g + (size_t)32 * NSZ], 0, 0);
            }
        }
        wait_asynccnt0();      // this wave's async copies landed in LDS
        __syncthreads();       // all waves' copies + ds stores visible block-wide

        // fragments per ISA 7.12.2:
        //  A (16x32 bf16): lane m, elems [0..7]=k 8h..8h+7, [8..15]=k 16+8h..16+8h+7
        //  B (32x16 bf16): lane n, elems [0..7]=k 8h..8h+7, [8..15]=k 16+8h..16+8h+7 (k-transposed LDS)
        const v8us* pAh = (const v8us*)sAh; const v8us* pAl = (const v8us*)sAl;
        const v8us* pBh = (const v8us*)sBh; const v8us* pBl = (const v8us*)sBl;
        Frag fAh[2], fAl[2], fBh[2], fBl[2];
#pragma unroll
        for (int i = 0; i < 2; ++i) {
            int row = wm + (i << 4) + mr;
            fAh[i].h[0] = pAh[(row << 2) + half];     fAh[i].h[1] = pAh[(row << 2) + 2 + half];
            fAl[i].h[0] = pAl[(row << 2) + half];     fAl[i].h[1] = pAl[(row << 2) + 2 + half];
        }
#pragma unroll
        for (int j = 0; j < 2; ++j) {
            int col = wn + (j << 4) + mr;
            fBh[j].h[0] = pBh[(col << 2) + half]; fBh[j].h[1] = pBh[(col << 2) + 2 + half];
            fBl[j].h[0] = pBl[(col << 2) + half]; fBl[j].h[1] = pBl[(col << 2) + 2 + half];
        }
#pragma unroll
        for (int i = 0; i < 2; ++i)
#pragma unroll
            for (int j = 0; j < 2; ++j) {
                acc[i][j] = wmma_bf16(fAh[i].bf, fBh[j].bf, acc[i][j]);  // hi*hi
                acc[i][j] = wmma_bf16(fAh[i].bf, fBl[j].bf, acc[i][j]);  // hi*lo
                acc[i][j] = wmma_bf16(fAl[i].bf, fBh[j].bf, acc[i][j]);  // lo*hi
            }
    }

    // epilogue: C/D layout — lane n=lane&15, VGPR v -> row v + 8*(lane>>4)
    size_t obase = (size_t)ch * NPIX;
#pragma unroll
    for (int i = 0; i < 2; ++i)
#pragma unroll
        for (int j = 0; j < 2; ++j) {
            int gn = n0 + wn + (j << 4) + mr;
#pragma unroll
            for (int v = 0; v < 8; ++v) {
                int gm = m0 + wm + (i << 4) + v + (half << 3);
                size_t off = obase + (size_t)gm * NSZ + gn;
                float f = acc[i][j][v];
                if (mode == 0) {
                    u16 hh, llv; split2(f, hh, llv);
                    oHi[off] = hh; oLo[off] = llv;
                } else {
                    oF[off] = fabsf(f);
                }
            }
        }
}

// -- percentile histogram on the 1/256 quantization grid
__global__ void k_pct_hist(const float* __restrict__ res, u32* __restrict__ hist) {
    int i = blockIdx.x * 256 + threadIdx.x;
    if (i >= NTOT) return;
    int c = i >> 18;
    int m = (int)(res[i] * 256.0f);   // res >= 0, trunc == floor
    if (m > PBINS - 1) m = PBINS - 1;
    if (m < 0) m = 0;
    atomicAdd(&hist[(size_t)c * PBINS + m], 1u);
}

__global__ void k_pct_select(const u32* __restrict__ hist, float* __restrict__ plo, float* __restrict__ phi) {
    if (threadIdx.x != 0) return;
    int c = blockIdx.x;
    const u32* h = hist + (size_t)c * PBINS;
    double rlo = 3.0 / 100.0 * (double)(NPIX - 1);
    double rhi = 97.0 / 100.0 * (double)(NPIX - 1);
    u32 jlo = (u32)rlo, jhi = (u32)rhi;
    float fl = (float)(rlo - (double)jlo), fh = (float)(rhi - (double)jhi);
    u32 targ[4] = { jlo, jlo + 1, jhi, jhi + 1 };
    float vals[4] = { 0.f, 0.f, 0.f, 0.f };
    u32 cum = 0; int ti = 0;
    for (int b = 0; b < PBINS && ti < 4; ++b) {
        u32 nc = cum + h[b];
        while (ti < 4 && targ[ti] < nc) { vals[ti] = (float)b * (1.0f / 256.0f); ++ti; }
        cum = nc;
    }
    while (ti < 4) { vals[ti] = (float)(PBINS - 1) * (1.0f / 256.0f); ++ti; }
    plo[c] = vals[0] + fl * (vals[1] - vals[0]);
    phi[c] = vals[2] + fh * (vals[3] - vals[2]);
}

__global__ void k_normalize(float* __restrict__ out, const float* __restrict__ mask,
                            const float* __restrict__ plo, const float* __restrict__ phi) {
    int i = blockIdx.x * 256 + threadIdx.x;
    if (i >= NTOT) return;
    int c = i >> 18;
    int b = i / 786432;
    int pix = i & (NPIX - 1);
    float L = plo[c], Hv = phi[c];
    out[i] = (out[i] - L) / (Hv - L) * mask[(size_t)b * NPIX + pix];
}

extern "C" void kernel_launch(void* const* d_in, const int* in_sizes, int n_in,
                              void* d_out, int out_size, void* d_ws, size_t ws_size,
                              hipStream_t stream) {
    const float* x    = (const float*)d_in[0];   // [32,3,512,512]
    const float* mask = (const float*)d_in[1];   // [32,1,512,512]
    const float* fmap = (const float*)d_in[2];   // [512,512]
    float* out = (float*)d_out;

    char* ws = (char*)d_ws;
    u16* Khi  = (u16*)(ws + OFF_KHI);
    u16* Klo  = (u16*)(ws + OFF_KLO);
    u16* K2hi = (u16*)(ws + OFF_K2HI);
    u16* K2lo = (u16*)(ws + OFF_K2LO);
    u16* Xhi  = (u16*)(ws + OFF_XHI);
    u16* Xlo  = (u16*)(ws + OFF_XLO);
    u16* Yhi  = (u16*)(ws + OFF_YHI);
    u16* Ylo  = (u16*)(ws + OFF_YLO);
    double* ku = (double*)(ws + OFF_KU);
    double* kv = (double*)(ws + OFF_KV);
    float* med = (float*)(ws + OFF_MED);
    u32* bucket = (u32*)(ws + OFF_BKT);
    u32* rem    = (u32*)(ws + OFF_REM);
    float* plo  = (float*)(ws + OFF_PLO);
    float* phi  = (float*)(ws + OFF_PHI);
    u32* hist   = (u32*)(ws + OFF_HIST);

    // convolution kernels from filter_map (separable exact reformulation of ifft2(fft2(x)*F))
    k_zero    <<<24576, 256, 0, stream>>>(hist, NCH * 65536);
    k_build_kd<<<2, 512, 0, stream>>>(fmap, ku, kv);
    k_build_K <<<1024, 256, 0, stream>>>(ku, kv, Khi, Klo, K2hi, K2lo);

    // exact per-channel median via 16+16-bit radix select
    k_hist_hi <<<98304, 256, 0, stream>>>(x, hist);
    k_select1 <<<96, 256, 0, stream>>>(hist, bucket, rem);
    k_hist_lo <<<98304, 256, 0, stream>>>(x, bucket, hist);
    k_select2 <<<96, 256, 0, stream>>>(hist, bucket, rem, med);

    // median padding + bf16 split
    k_split_x <<<98304, 256, 0, stream>>>(x, mask, med, Xhi, Xlo);

    // Y1 = K_left @ Xp   (per channel), split-bf16 output
    k_gemm<<<NCH * 32, 256, 0, stream>>>(Khi, Klo, (size_t)0, Xhi, Xlo, (size_t)NPIX,
                                         Yhi, Ylo, nullptr, 0);
    // Y2 = |Y1 @ K_right| -> d_out (f32)
    k_gemm<<<NCH * 32, 256, 0, stream>>>(Yhi, Ylo, (size_t)NPIX, K2hi, K2lo, (size_t)0,
                                         nullptr, nullptr, out, 1);

    // percentile normalization (exact on the 1/256 quantization grid) + mask
    k_pct_hist  <<<98304, 256, 0, stream>>>(out, hist);
    k_pct_select<<<96, 64, 0, stream>>>(hist, plo, phi);
    k_normalize <<<98304, 256, 0, stream>>>(out, mask, plo, phi);
}
